// GAP_12240656793836
// MI455X (gfx1250) — matrix operand order repaired and checked by reading, not compiled, not run on hardware
//
#include <hip/hip_runtime.h>
#include <hip/hip_bf16.h>

// ---------------------------------------------------------------------------
// GAP bilinear-attention pooling for MI455X (gfx1250), wave32 + WMMA f16.
// One workgroup per batch element; 256 threads = 8 waves.
// Round 3: 4-wide vectorized embedding gather (float4 loads -> ds_store_b64),
// neighbor indices staged in LDS, global_prefetch warm-up of Wt16 rows.
// ---------------------------------------------------------------------------

typedef __attribute__((ext_vector_type(16))) _Float16 v16h;
typedef __attribute__((ext_vector_type(4)))  _Float16 v4h;
typedef __attribute__((ext_vector_type(8)))  float    v8f;

namespace {
constexpr int D     = 256;       // embedding dim
constexpr int Lnb   = 100;       // neighborhood length
constexpr int LP    = 112;       // padded to multiple of 16
constexpr int LDH   = 264;       // f16 row stride (528B rows -> conflict-free A loads)
constexpr int SIMLD = 116;       // f32 row stride for sim
constexpr int MT    = LP / 16;   // 7 M tiles
constexpr int KST   = D / 32;    // 8 k-steps of 32

// LDS layout (bytes)
constexpr int OFF_S   = 0;
constexpr int OFF_T   = OFF_S   + LP * LDH * 2;     //  59,136
constexpr int OFF_P   = OFF_T   + LP * LDH * 2;     // 118,272
constexpr int OFF_SIM = OFF_P   + LP * LDH * 2;     // 177,408
constexpr int OFF_AS  = OFF_SIM + LP * SIMLD * 4;   // 229,376
constexpr int OFF_AT  = OFF_AS  + LP * 4;
constexpr int LDS_BYTES = OFF_AT + LP * 4;          // 230,272  (< 320K WGP LDS)
}

__device__ __forceinline__ v8f zero8() {
  v8f z = {0.f, 0.f, 0.f, 0.f, 0.f, 0.f, 0.f, 0.f};
  return z;
}

// A-matrix 16x32 f16 tile (row-major, stride ld), contiguous k order, same as the B loader:
// lane L holds row M = L%16; lanes 0-15 K=0..15, lanes 16-31 K=16..31.
__device__ __forceinline__ v16h load_A16x32(const _Float16* base, int m0, int k0, int ld) {
  const int lane  = threadIdx.x & 31;
  const int m     = lane & 15;
  const int khalf = (lane >> 4) << 4;   // 0 or 16
  const _Float16* rp = base + (m0 + m) * ld + k0 + khalf;
  v16h a;
#pragma unroll
  for (int v = 0; v < 8; ++v) {
    const int k = v << 1;
    a[2 * v]     = rp[k];
    a[2 * v + 1] = rp[k + 1];
  }
  return a;
}

// B-matrix 32x16 f16 tile where B[k][n] = src[(n0+n)*ld + k0+k] (src holds the
// transposed operand row-major, so each lane reads 16 contiguous f16 = 32B).
// Layout: lane holds column N = lane%16; lanes 0-15 K=0..15, lanes 16-31 K=16..31.
__device__ __forceinline__ v16h load_B_from_T(const _Float16* src, int n0, int k0, int ld) {
  const int lane = threadIdx.x & 31;
  const int n    = lane & 15;
  const int kb   = (lane >> 4) << 4;    // 0 or 16
  const _Float16* rp = src + (n0 + n) * ld + k0 + kb;
  v16h b;
#pragma unroll
  for (int v = 0; v < 8; ++v) {
    b[2 * v]     = rp[2 * v];
    b[2 * v + 1] = rp[2 * v + 1];
  }
  return b;
}

// ---------------------------------------------------------------------------
// Prep: Wt16[e][d] = f16(W[d][e])  (128 KB, stays hot in the 192 MB L2)
// ---------------------------------------------------------------------------
__global__ __launch_bounds__(256)
void convert_W_kernel(const float* __restrict__ W, _Float16* __restrict__ Wt) {
  const int e = blockIdx.x;    // output row   (0..255)
  const int d = threadIdx.x;   // output col   (0..255)
  Wt[e * D + d] = (_Float16)W[d * D + e];
}

// ---------------------------------------------------------------------------
// Main kernel
// ---------------------------------------------------------------------------
__global__ __launch_bounds__(256)
void gap_wmma_kernel(const int*      __restrict__ src_idx,
                     const int*      __restrict__ tgt_idx,
                     const float*    __restrict__ src_mask,
                     const float*    __restrict__ tgt_mask,
                     const float*    __restrict__ emb,
                     const _Float16* __restrict__ Wt16,   // [e][d], transposed f16
                     float*          __restrict__ out,
                     int batches) {
  extern __shared__ char smem[];
  _Float16* s_h   = (_Float16*)(smem + OFF_S);
  _Float16* t_h   = (_Float16*)(smem + OFF_T);
  _Float16* p_h   = (_Float16*)(smem + OFF_P);
  float*    sim   = (float*)   (smem + OFF_SIM);
  float*    att_s = (float*)   (smem + OFF_AS);
  float*    att_t = (float*)   (smem + OFF_AT);

  const int b    = blockIdx.x;
  const int tid  = threadIdx.x;
  const int lane = tid & 31;
  const int wv   = tid >> 5;             // 8 waves

  // Warm the WGP$ with the Wt16 rows this wave's two GEMM1 N-tiles will read
  // (rows 32*wv .. 32*wv+31, 512B each) while the gather is in flight.
  {
    const _Float16* wrow = Wt16 + (wv * 32 + lane) * D;
    __builtin_prefetch(wrow, 0, 3);
    __builtin_prefetch(wrow + 128, 0, 3);
  }

  // ---- Stage 0a: stage neighbor indices in LDS (att_* scratch is dead until
  //      stage 3; padded rows get index 0 -> zero embedding) ------------------
  int* si = (int*)att_s;
  int* ti = (int*)att_t;
  if (tid < LP) {
    si[tid] = (tid < Lnb) ? src_idx[b * Lnb + tid] : 0;
    ti[tid] = (tid < Lnb) ? tgt_idx[b * Lnb + tid] : 0;
  }
  __syncthreads();

  // ---- Stage 0b: gather embeddings fp32 -> f16 into LDS, 4 dims/thread ------
  // Thread covers dims [4*(tid%64), +4) of row (r0 + tid/64):
  // one global_load_b128 + two cvt_pk + one ds_store_b64 per matrix per iter.
  {
    const int dsub = (tid & 63) << 2;    // 0..252
    const int rsub = tid >> 6;           // 0..3
    for (int r0 = 0; r0 < LP; r0 += 4) {
      const int r  = r0 + rsub;
      const int is = si[r];
      const int it = ti[r];
      float4 vs = {0.f, 0.f, 0.f, 0.f};
      float4 vt = {0.f, 0.f, 0.f, 0.f};
      if (is != 0) vs = *(const float4*)(emb + (long long)is * D + dsub);
      if (it != 0) vt = *(const float4*)(emb + (long long)it * D + dsub);
      v4h hs, ht;
      hs[0] = (_Float16)vs.x; hs[1] = (_Float16)vs.y;
      hs[2] = (_Float16)vs.z; hs[3] = (_Float16)vs.w;
      ht[0] = (_Float16)vt.x; ht[1] = (_Float16)vt.y;
      ht[2] = (_Float16)vt.z; ht[3] = (_Float16)vt.w;
      *(v4h*)(s_h + r * LDH + dsub) = hs;   // 8B aligned: 528*r + 8*(dsub/4)
      *(v4h*)(t_h + r * LDH + dsub) = ht;
    }
  }
  __syncthreads();

  // ---- Stage 1 (GEMM1): P[112,256] = s[112,256] x W[256,256] ----------------
  // Each wave owns 2 N-tile columns; f16 B tile from L2-resident Wt16 hoisted
  // over the 7 M tiles: two global_load_b128 per lane per k-step.
  {
    for (int nn = 0; nn < 2; ++nn) {
      const int n0 = (wv * 2 + nn) * 16;
      v8f acc[MT];
#pragma unroll
      for (int m = 0; m < MT; ++m) acc[m] = zero8();

      for (int ks = 0; ks < KST; ++ks) {
        const int k0 = ks * 32;
        v16h btile = load_B_from_T(Wt16, n0, k0, D);   // B[k][n] = W[k][n]
#pragma unroll
        for (int m = 0; m < MT; ++m) {
          v16h a = load_A16x32(s_h, m * 16, k0, LDH);
          acc[m] = __builtin_amdgcn_wmma_f32_16x16x32_f16(
              false, a, false, btile, (short)0, acc[m], false, false);
        }
      }
      // Store C (fp32) -> P (f16). C layout: VGPR r -> M = r + 8*(lane>>4).
      {
        const int n  = lane & 15;
        const int mo = (lane >> 4) << 3;
#pragma unroll
        for (int m = 0; m < MT; ++m)
#pragma unroll
          for (int r = 0; r < 8; ++r)
            p_h[(m * 16 + mo + r) * LDH + n0 + n] = (_Float16)acc[m][r];
      }
    }
  }
  __syncthreads();

  // ---- Stage 2 (GEMM2): sim[112,112] = tanh(P[112,256] x tg[112,256]^T) -----
  for (int t = wv; t < MT * MT; t += 8) {
    const int m0 = (t / MT) * 16;
    const int n0 = (t % MT) * 16;
    v8f acc = zero8();
    for (int ks = 0; ks < KST; ++ks) {
      const int k0 = ks * 32;
      v16h a = load_A16x32(p_h, m0, k0, LDH);
      v16h btile = load_B_from_T(t_h, n0, k0, LDH);   // B[k][n] = tg[n][k]
      acc = __builtin_amdgcn_wmma_f32_16x16x32_f16(
          false, a, false, btile, (short)0, acc, false, false);
    }
    const int n  = lane & 15;
    const int mo = (lane >> 4) << 3;
#pragma unroll
    for (int r = 0; r < 8; ++r)
      sim[(m0 + mo + r) * SIMLD + n0 + n] = tanhf(acc[r]);
  }
  __syncthreads();

  // ---- Stage 3: row/col means + masks (padded slots -> -inf) ----------------
  if (tid < LP) {
    float rs = 0.f, cs = 0.f;
    for (int c = 0; c < LP; ++c) rs += sim[tid * SIMLD + c];   // padded cols == 0
    for (int r = 0; r < LP; ++r) cs += sim[r * SIMLD + tid];   // padded rows == 0
    att_s[tid] = (tid < Lnb) ? (rs * (1.0f / Lnb) + src_mask[b * Lnb + tid]) : -1e30f;
    att_t[tid] = (tid < Lnb) ? (cs * (1.0f / Lnb) + tgt_mask[b * Lnb + tid]) : -1e30f;
  }
  __syncthreads();

  // ---- Stage 4: softmax (wave 0 -> att_s, wave 1 -> att_t), wave32 shuffles -
  if (wv < 2) {
    float* att = wv ? att_t : att_s;
    float mx = -1e30f;
    for (int i = lane; i < Lnb; i += 32) mx = fmaxf(mx, att[i]);
#pragma unroll
    for (int o = 16; o >= 1; o >>= 1) mx = fmaxf(mx, __shfl_xor(mx, o, 32));
    float sum = 0.f;
    for (int i = lane; i < Lnb; i += 32) {
      const float e = __expf(att[i] - mx);
      att[i] = e;
      sum += e;
    }
#pragma unroll
    for (int o = 16; o >= 1; o >>= 1) sum += __shfl_xor(sum, o, 32);
    const float inv = 1.0f / sum;
    for (int i = lane; i < Lnb; i += 32) att[i] *= inv;
  }
  __syncthreads();

  // ---- Stage 5: attention-weighted reductions, thread == dim ----------------
  {
    float accs = 0.f, acct = 0.f;
    for (int s = 0; s < Lnb; ++s) {
      accs += (float)s_h[s * LDH + tid] * att_s[s];
      acct += (float)t_h[s * LDH + tid] * att_t[s];
    }
    out[(long long)b * D + tid] = accs;                                   // source_rep
    out[(long long)batches * D + (long long)b * D + tid] = acct;          // target_rep
  }
}

extern "C" void kernel_launch(void* const* d_in, const int* in_sizes, int n_in,
                              void* d_out, int out_size, void* d_ws, size_t ws_size,
                              hipStream_t stream) {
  const int*   src_idx  = (const int*)  d_in[0];
  const int*   tgt_idx  = (const int*)  d_in[1];
  const float* src_mask = (const float*)d_in[2];
  const float* tgt_mask = (const float*)d_in[3];
  const float* emb      = (const float*)d_in[4];
  const float* W        = (const float*)d_in[5];
  float*       out      = (float*)d_out;
  _Float16*    Wt16     = (_Float16*)d_ws;         // 256*256*2 = 128 KB scratch

  const int batches = in_sizes[0] / Lnb;           // 1024

  // Dynamic LDS > default opt-in (deterministic, capture-safe host call).
  hipFuncSetAttribute((const void*)gap_wmma_kernel,
                      hipFuncAttributeMaxDynamicSharedMemorySize, LDS_BYTES);

  convert_W_kernel<<<D, 256, 0, stream>>>(W, Wt16);
  gap_wmma_kernel<<<batches, 256, LDS_BYTES, stream>>>(
      src_idx, tgt_idx, src_mask, tgt_mask, emb, Wt16, out, batches);
}
